// MAF_3358664426273
// MI455X (gfx1250) — matrix operand order repaired and checked by reading, not compiled, not run on hardware
//
#include <hip/hip_runtime.h>
#include <cmath>

// ---- MI455X / gfx1250: wave32, WMMA f32_16x16x32_f16 ----
typedef _Float16 f16x2  __attribute__((ext_vector_type(2)));
typedef _Float16 f16x8  __attribute__((ext_vector_type(8)));
typedef _Float16 f16x16 __attribute__((ext_vector_type(16)));
typedef float    f32x8  __attribute__((ext_vector_type(8)));

#define BATCH 16384
#define DIM   64
#define HID   128
#define NLAY  63
#define BT    128          // batch rows per workgroup

// LDS strides (halves). x2 bytes must be multiple of 16 for b128 frag loads.
#define SXLD  72           // 64 + 8  -> 144 B row
#define SW1LD 72           // K<=64   -> 144 B row (transposed: [n][k])
#define SW2LD 136          // 128 + 8 -> 272 B row (transposed: [n][k])
#define SH1LD 136          // 128 + 8 -> 272 B row ([m][k])

static __device__ __forceinline__ f16x16 cat8(f16x8 lo, f16x8 hi) {
  return __builtin_shufflevector(lo, hi, 0,1,2,3,4,5,6,7,8,9,10,11,12,13,14,15);
}

static __device__ __forceinline__ float fast_rcp(float x) {
#if __has_builtin(__builtin_amdgcn_rcpf)
  return __builtin_amdgcn_rcpf(x);
#else
  return 1.0f / x;
#endif
}

static __device__ __forceinline__ float fast_exp2(float x) {
#if __has_builtin(__builtin_amdgcn_exp2f)
  return __builtin_amdgcn_exp2f(x);
#else
  return exp2f(x);
#endif
}

// Branch-free tanh: hardware v_tanh_f32 if the builtin exists on this
// toolchain, else 1 - 2/(exp2(2*log2e*x)+1) via v_exp_f32/v_rcp_f32.
// Saturates correctly: x>>0 -> exp2->inf -> rcp->0 -> 1; x<<0 -> -1.
static __device__ __forceinline__ float fast_tanh(float x) {
#if __has_builtin(__builtin_amdgcn_tanhf)
  return __builtin_amdgcn_tanhf(x);
#elif __has_builtin(__builtin_amdgcn_tanh_f32)
  return __builtin_amdgcn_tanh_f32(x);
#else
  float e = fast_exp2(x * 2.8853900817779268f);  // 2*log2(e)
  return 1.0f - 2.0f * fast_rcp(e + 1.0f);
#endif
}

static __device__ __forceinline__ float fast_expneg(float a) {
  return fast_exp2(a * -1.4426950408889634f);    // exp(-a)
}

static __device__ __forceinline__ f16x2 pk_f16(float a, float b) {
#if __has_builtin(__builtin_amdgcn_cvt_pkrtz)
  // builtin returns __fp16 ext_vector(2); same bits as f16x2 -> bit_cast
  return __builtin_bit_cast(f16x2, __builtin_amdgcn_cvt_pkrtz(a, b));
#else
  f16x2 r; r.x = (_Float16)a; r.y = (_Float16)b; return r;
#endif
}

__global__ __launch_bounds__(256)
void maf_layer_kernel(const float* __restrict__ x,
                      const float* __restrict__ W1, const float* __restrict__ b1,
                      const float* __restrict__ W2, const float* __restrict__ b2,
                      const float* __restrict__ W3, const float* __restrict__ b3,
                      float* __restrict__ zout,      // [B,64] region of d_out
                      float* __restrict__ alpha_ws)  // [B,64] scratch
{
  const int l    = blockIdx.y;          // layer 0..62  -> produces dim l+1
  const int m0   = blockIdx.x * BT;     // batch-row base
  const int tid  = threadIdx.x;
  const int wave = tid >> 5;            // 0..7
  const int lane = tid & 31;
  const int l16  = lane & 15;
  const int lh   = lane >> 4;           // 0/1 half-group

  __shared__ __align__(16) _Float16 sX  [BT  * SXLD ];   // A1: [m][k<64]
  __shared__ __align__(16) _Float16 sW1t[HID * SW1LD];   // B1^T: [n][k<64]
  __shared__ __align__(16) _Float16 sW2t[HID * SW2LD];   // B2^T: [n][k<128]
  __shared__ __align__(16) _Float16 sH1 [BT  * SH1LD];   // A2: [m][k<128]
  __shared__ float sB1[HID], sB2[HID], sW3a[HID], sW3b[HID], sB3[2];

  // ---- stage inputs / weights as f16 pairs (packed converts + b32 stores) ----
  // x tile: pairs along columns
  for (int idx = tid; idx < BT * (DIM / 2); idx += 256) {
    int r = idx >> 5, c2 = (idx & 31) * 2;
    float2 v = *(const float2*)&x[(size_t)(m0 + r) * DIM + c2];
    *(f16x2*)&sX[r * SXLD + c2] = pk_f16(v.x, v.y);
  }
  // W1^T (triangular mask, K padded to 64): pairs along k
  for (int idx = tid; idx < HID * 32; idx += 256) {
    int n = idx & 127, kp = idx >> 7;       // kp = 0..31 -> k = 2kp, 2kp+1
    int k0 = 2 * kp, k1 = 2 * kp + 1;
    float v0 = (k0 <= l && k0 < 63) ? W1[((size_t)l * 63 + k0) * HID + n] : 0.0f;
    float v1 = (k1 <= l && k1 < 63) ? W1[((size_t)l * 63 + k1) * HID + n] : 0.0f;
    *(f16x2*)&sW1t[n * SW1LD + k0] = pk_f16(v0, v1);
  }
  // W2^T: pairs along k; consecutive tid -> consecutive n -> coalesced loads
  for (int idx = tid; idx < HID * 64; idx += 256) {
    int n = idx & 127, kp = idx >> 7;       // kp = 0..63 -> k = 2kp, 2kp+1
    int k0 = 2 * kp;
    float v0 = W2[((size_t)l * HID + k0) * HID + n];
    float v1 = W2[((size_t)l * HID + k0 + 1) * HID + n];
    *(f16x2*)&sW2t[n * SW2LD + k0] = pk_f16(v0, v1);
  }
  if (tid < HID) {
    sB1 [tid] = b1[(size_t)l * HID + tid];
    sB2 [tid] = b2[(size_t)l * HID + tid];
    sW3a[tid] = W3[((size_t)l * HID + tid) * 2 + 0];
    sW3b[tid] = W3[((size_t)l * HID + tid) * 2 + 1];
  }
  if (tid < 2) sB3[tid] = b3[(size_t)l * 2 + tid];
  __syncthreads();

  const int rowA = 16 * wave + l16;     // A-frag row for this lane

  // ---- GEMM1: h1 = tanh(X[128x64] * W1[64x128] + b1) ----
  for (int tn = 0; tn < 8; ++tn) {
    f32x8 acc = {};
    #pragma unroll
    for (int kk = 0; kk < 64; kk += 32) {
      const _Float16* ap = &sX[rowA * SXLD + kk + 8 * lh];
      f16x16 a = cat8(*(const f16x8*)ap, *(const f16x8*)(ap + 16));
      const _Float16* bp = &sW1t[(16 * tn + l16) * SW1LD + kk + 8 * lh];
      f16x16 b = cat8(*(const f16x8*)bp, *(const f16x8*)(bp + 16));
      acc = __builtin_amdgcn_wmma_f32_16x16x32_f16(false, a, false, b,
                                                   (short)0, acc, false, false);
    }
    const int n = 16 * tn + l16;
    const float bias = sB1[n];
    #pragma unroll
    for (int r = 0; r < 8; ++r) {
      int m = 16 * wave + 8 * lh + r;   // D-matrix row for VGPR r
      sH1[m * SH1LD + n] = (_Float16)fast_tanh(acc[r] + bias);
    }
  }
  __syncthreads();

  // ---- GEMM2 + fused GEMM3 partials: h2 = tanh(h1*W2+b2); p = h2*W3 ----
  float p0[8] = {}, p1[8] = {};
  for (int tn = 0; tn < 8; ++tn) {
    f32x8 acc = {};
    #pragma unroll
    for (int kk = 0; kk < 128; kk += 32) {
      const _Float16* ap = &sH1[rowA * SH1LD + kk + 8 * lh];
      f16x16 a = cat8(*(const f16x8*)ap, *(const f16x8*)(ap + 16));
      const _Float16* bp = &sW2t[(16 * tn + l16) * SW2LD + kk + 8 * lh];
      f16x16 b = cat8(*(const f16x8*)bp, *(const f16x8*)(bp + 16));
      acc = __builtin_amdgcn_wmma_f32_16x16x32_f16(false, a, false, b,
                                                   (short)0, acc, false, false);
    }
    const int n = 16 * tn + l16;
    const float bias = sB2[n], w3a = sW3a[n], w3b = sW3b[n];
    #pragma unroll
    for (int r = 0; r < 8; ++r) {
      float h2 = fast_tanh(acc[r] + bias);
      p0[r] += h2 * w3a;
      p1[r] += h2 * w3b;
    }
  }

  // ---- reduce over the 16-lane N-groups (wave32 shuffles) ----
  #pragma unroll
  for (int r = 0; r < 8; ++r) {
    #pragma unroll
    for (int m = 8; m >= 1; m >>= 1) {
      p0[r] += __shfl_xor(p0[r], m, 32);
      p1[r] += __shfl_xor(p1[r], m, 32);
    }
  }

  // ---- epilogue: mu/alpha for dim l+1 -> z column (62-l), stash alpha ----
  if (l16 == 0) {   // lanes 0 and 16: rows 16w+0..7 / 16w+8..15
    #pragma unroll
    for (int r = 0; r < 8; ++r) {
      int row = m0 + 16 * wave + 8 * lh + r;
      float mu = p0[r] + sB3[0];
      float al = p1[r] + sB3[1];
      float xv = x[(size_t)row * DIM + (l + 1)];
      zout[(size_t)row * DIM + (62 - l)]    = (xv - mu) * fast_expneg(al);
      alpha_ws[(size_t)row * DIM + (l + 1)] = al;
    }
  }
}

__global__ __launch_bounds__(256)
void maf_finalize_kernel(const float* __restrict__ x,
                         const float* __restrict__ initial_param,
                         const float* __restrict__ alpha_ws,
                         float* __restrict__ out)   // [B,64] z_rev, then [B] log_det
{
  int row = blockIdx.x * blockDim.x + threadIdx.x;
  if (row >= BATCH) return;
  const float mu0 = initial_param[0];
  const float a0  = initial_param[1];
  // dim 0 -> reversed position 63
  out[(size_t)row * DIM + 63] = (x[(size_t)row * DIM] - mu0) * fast_expneg(a0);
  // deterministic fixed-order log-det reduction
  float ld = -a0;
  for (int i = 1; i < DIM; ++i) ld -= alpha_ws[(size_t)row * DIM + i];
  out[(size_t)BATCH * DIM + row] = ld;
}

extern "C" void kernel_launch(void* const* d_in, const int* in_sizes, int n_in,
                              void* d_out, int out_size, void* d_ws, size_t ws_size,
                              hipStream_t stream) {
  const float* x   = (const float*)d_in[0];
  const float* W1  = (const float*)d_in[1];
  const float* b1  = (const float*)d_in[2];
  const float* W2  = (const float*)d_in[3];
  const float* b2  = (const float*)d_in[4];
  const float* W3  = (const float*)d_in[5];
  const float* b3  = (const float*)d_in[6];
  const float* ip  = (const float*)d_in[7];
  float* out      = (float*)d_out;
  float* alpha_ws = (float*)d_ws;     // needs B*64*4 = 4 MB scratch

  dim3 grid(BATCH / BT, NLAY);
  maf_layer_kernel<<<grid, 256, 0, stream>>>(x, W1, b1, W2, b2, W3, b3,
                                             out, alpha_ws);
  maf_finalize_kernel<<<BATCH / 256, 256, 0, stream>>>(x, ip, alpha_ws, out);
}
